// ConvGeodesic_4294967296013
// MI455X (gfx1250) — matrix operand order repaired and checked by reading, not compiled, not run on hardware
//
#include <hip/hip_runtime.h>
#include <hip/hip_bf16.h>

// ---------------------------------------------------------------------------
// ConvGeodesic on MI455X (gfx1250, wave32, WMMA).
//
// out[b,n,g,o] = sum_{r,a,f} interp[b,g,r,a,f] * W[r,(a+n)%8,o,f] + 2*sum_{r,a} bias
// with interp = 3-point barycentric gather-blend of signal.
// Framed as GEMM: M = B*G = 48000, Ncols = NROT*O = 256, Kdim = R*A*F = 640.
// f16 operands, f32 accumulation via v_wmma_f32_16x16x32_f16.
// ---------------------------------------------------------------------------

#define BB   4
#define VV   12000
#define FF   16
#define GG   12000
#define RR   5
#define AA   8
#define OO   32
#define NROT 8

#define KDIM   640          // R*A*F
#define COLS   256          // NROT*O
#define ROWTOT 48000        // B*G

#define ROWS_PER_WG 128
#define COLS_PER_WG 64
#define KSTEP       32
#define NKSTEP      (KDIM / KSTEP)   // 20

typedef __attribute__((ext_vector_type(16))) _Float16    v16h;
typedef __attribute__((ext_vector_type(8)))  float       v8f;
typedef __attribute__((ext_vector_type(4)))  unsigned    v4u;

// LDS layout (dynamic extern shared; 245,888 B < 320 KB per WGP on CDNA5)
#define SA_ELEMS (ROWS_PER_WG * KDIM)          // 128*640 f16 = 160 KB
#define SB_ELEMS (COLS_PER_WG * KDIM)          // 64*640 f16  =  80 KB
#define SA_BYTES (SA_ELEMS * 2)
#define SB_BYTES (SB_ELEMS * 2)
#define SHMEM_BYTES (SA_BYTES + SB_BYTES + OO * 4)

__global__ __launch_bounds__(256)
void conv_geodesic_wmma(const float* __restrict__ signal,
                        const float* __restrict__ bary,
                        const float* __restrict__ kernels,
                        const float* __restrict__ bias,
                        float* __restrict__ out) {
    extern __shared__ char smem[];
    _Float16* sA = (_Float16*)smem;                         // [row][c]  row-major, stride KDIM
    _Float16* sB = (_Float16*)(smem + SA_BYTES);            // [col][c]  row-major, stride KDIM
    float*    bs = (float*)(smem + SA_BYTES + SB_BYTES);    // [32] bias epilogue

    const int tid     = threadIdx.x;
    const int cb      = blockIdx.x;            // column tile: cols [cb*64, cb*64+64)
    const int rowBase = blockIdx.y * ROWS_PER_WG;

    // ---- Phase 0: bias epilogue: bs[o] = K * sum_{r,a} bias[r,a,o] ----------
    if (tid < OO) {
        float s = 0.f;
        #pragma unroll
        for (int ra = 0; ra < RR * AA; ++ra) s += bias[ra * OO + tid];
        bs[tid] = 2.0f * s;                    // K = 2
    }

    // ---- Phase 1: build rotated+k-summed weight slice into sB (f16) ---------
    // sB[j][c] = sum_k kernels[k, r, (a+n)%8, o, f], c=(r*8+a)*16+f, col=cb*64+j=(n,o)
    for (int it = tid; it < COLS_PER_WG * (KDIM / 2); it += 256) {
        const int j   = it / (KDIM / 2);
        const int cp  = it - j * (KDIM / 2);
        const int c   = cp * 2;                // even; (c, c+1) share r,a and f,f+1
        const int r   = c >> 7;                // c / 128
        const int rem = c & 127;
        const int a   = rem >> 4;
        const int f   = rem & 15;
        const int col = cb * COLS_PER_WG + j;
        const int n   = col >> 5;
        const int o   = col & 31;
        const int a2  = (a + n) & 7;
        const int kb  = ((r * AA + a2) * OO + o) * FF + f;   // kernels[0,...]
        const float v0 = kernels[kb]     + kernels[kb + RR * AA * OO * FF];
        const float v1 = kernels[kb + 1] + kernels[kb + 1 + RR * AA * OO * FF];
        union { _Float16 h[2]; unsigned u; } pk;
        pk.h[0] = (_Float16)v0;
        pk.h[1] = (_Float16)v1;
        ((unsigned*)sB)[j * (KDIM / 2) + cp] = pk.u;
    }

    // ---- Phase 2: gather + barycentric blend -> sA (f16) --------------------
    // sA[i][ra*16+f] = sum_{s=0..2} w_s * signal[b, idx_s, f]
    for (int task = tid; task < ROWS_PER_WG * RR * AA; task += 256) {
        const int i   = task / (RR * AA);
        const int ra  = task - i * (RR * AA);
        const int row = rowBase + i;
        const int b   = row / GG;
        const int g   = row - b * GG;
        const float* bp = bary + (size_t)row * (RR * AA * 6) + ra * 6;
        const float w0 = bp[1], w1 = bp[3], w2 = bp[5];
        const int   i0 = (int)bp[0], i1 = (int)bp[2], i2 = (int)bp[4];
        const float* s0 = signal + ((size_t)b * VV + i0) * FF;
        const float* s1 = signal + ((size_t)b * VV + i1) * FF;
        const float* s2 = signal + ((size_t)b * VV + i2) * FF;
        union { _Float16 h[16]; v4u q[2]; } u;
        #pragma unroll
        for (int f = 0; f < FF; ++f) {
            u.h[f] = (_Float16)(w0 * s0[f] + w1 * s1[f] + w2 * s2[f]);
        }
        v4u* dst = (v4u*)&sA[i * KDIM + ra * FF];
        dst[0] = u.q[0];
        dst[1] = u.q[1];
    }

    __syncthreads();

    // ---- Phase 3: WMMA GEMM. Wave w: rows [16w,16w+16), all 64 cols ---------
    const int wave   = tid >> 5;
    const int lane   = tid & 31;
    const int laneM  = lane & 15;     // M for A / N for B / N for D
    const int laneHi = lane >> 4;     // lane-half select

    v8f acc[4] = {};

    // A fragment (16x32 f16, ISA 7.12.2): elements 0-7 at c = laneHi*8 + {0..7},
    // elements 8-15 at c = 16 + laneHi*8 + {0..7}   -> two b128 LDS loads.
    const int aBase = (wave * 16 + laneM) * KDIM + laneHi * 8;
    // B fragment (32x16 f16): same k order as A: elements 0-7 at c = laneHi*8 + {0..7},
    // elements 8-15 at c = 16 + laneHi*8 + {0..7}   -> two b128 LDS loads.
    const int bBase = laneM * KDIM + laneHi * 8;

    #pragma unroll 2
    for (int t = 0; t < NKSTEP; ++t) {
        union { v16h h; v4u q[2]; } af;
        const v4u* ap = (const v4u*)&sA[aBase + t * KSTEP];
        af.q[0] = ap[0];          // f16 offsets 0..7
        af.q[1] = ap[2];          // f16 offsets 16..23
        #pragma unroll
        for (int jt = 0; jt < 4; ++jt) {
            union { v16h h; v4u q[2]; } bf;
            const v4u* bp2 = (const v4u*)&sB[jt * 16 * KDIM + bBase + t * KSTEP];
            bf.q[0] = bp2[0];
            bf.q[1] = bp2[2];
            acc[jt] = __builtin_amdgcn_wmma_f32_16x16x32_f16(
                false, af.h, false, bf.h, (short)0, acc[jt], false, false);
        }
    }

    // ---- Phase 4: epilogue + scatter to out[b,n,g,o] ------------------------
    // D layout: lane -> N = laneM; VGPR e -> M = e + 8*laneHi.
    #pragma unroll
    for (int jt = 0; jt < 4; ++jt) {
        const int col = cb * COLS_PER_WG + jt * 16 + laneM;
        const int n   = col >> 5;
        const int o   = col & 31;
        const float bo = bs[o];
        #pragma unroll
        for (int e = 0; e < 8; ++e) {
            const int M   = e + laneHi * 8;
            const int row = rowBase + wave * 16 + M;
            const int b   = row / GG;
            const int g   = row - b * GG;
            out[(((size_t)(b * NROT + n)) * GG + g) * OO + o] = acc[jt][e] + bo;
        }
    }
}

extern "C" void kernel_launch(void* const* d_in, const int* in_sizes, int n_in,
                              void* d_out, int out_size, void* d_ws, size_t ws_size,
                              hipStream_t stream) {
    (void)in_sizes; (void)n_in; (void)d_ws; (void)ws_size; (void)out_size;
    const float* signal  = (const float*)d_in[0];
    const float* bary    = (const float*)d_in[1];
    const float* kernels = (const float*)d_in[2];
    const float* bias    = (const float*)d_in[3];
    // d_in[4] = rotation_delta (==1 in setup): NROT = A hardcoded accordingly.
    float* out = (float*)d_out;

    dim3 grid(COLS / COLS_PER_WG, ROWTOT / ROWS_PER_WG);   // (4, 375)
    dim3 block(256);                                        // 8 wave32 waves
    conv_geodesic_wmma<<<grid, block, SHMEM_BYTES, stream>>>(
        signal, bary, kernels, bias, out);
}
